// CoCNN_67525475827831
// MI455X (gfx1250) — matrix operand order repaired and checked by reading, not compiled, not run on hardware
//
#include <hip/hip_runtime.h>
#include <hip/hip_bf16.h>

typedef __attribute__((ext_vector_type(16))) _Float16 v16h;
typedef __attribute__((ext_vector_type(8)))  _Float16 v8h;
typedef __attribute__((ext_vector_type(8)))  float    v8f;

#define CDIV(a,b) (((a)+(b)-1)/(b))

__device__ __forceinline__ v16h ld16(const _Float16* p) {
  const v8h* q = (const v8h*)p;
  v8h a = q[0], b = q[1];
  return __builtin_shufflevector(a, b, 0,1,2,3,4,5,6,7,8,9,10,11,12,13,14,15);
}

// ---------------------------------------------------------------------------
// Weight packing: f32 OIHW -> f16 in WMMA A-fragment order.
// Per (ocTile mt, tap t, ciTile kt): 32 lanes x 16 halfs; lane l element e is
// wt[oc = mt*16 + (l&15), ci = kt*32 + klocal, t],
// klocal = e + 16*(l>>4)   (contiguous k order, matches B loads).
// permSx>0 (FC1): T==1 and ref k = (ci % permCx)*permSx + ci/permCx.
// ---------------------------------------------------------------------------
__global__ void pack_w_kernel(const float* __restrict__ wt, _Float16* __restrict__ out,
                              int Cout, int Cin, int T, int Cinp, int permSx, int permCx,
                              int total)
{
  int i = blockIdx.x * blockDim.x + threadIdx.x;
  if (i >= total) return;
  const int KT = Cinp >> 5;
  int e = i & 15;
  int l = (i >> 4) & 31;
  int rest = i >> 9;                 // ((mt*T + t)*KT + kt)
  int kt = rest % KT; rest /= KT;
  int t  = rest % T;
  int mt = rest / T;
  int half = l >> 4, m = l & 15;
  int klocal = e + half * 16;
  int ci = kt * 32 + klocal;
  int oc = mt * 16 + m;
  float v = 0.0f;
  if (oc < Cout && ci < Cin) {
    if (permSx > 0) {
      int rk = (ci % permCx) * permSx + ci / permCx;
      v = wt[(size_t)oc * Cin + rk];
    } else {
      v = wt[((size_t)oc * Cin + ci) * T + t];
    }
  }
  out[i] = (_Float16)v;
}

// NCHW f32 -> NHWC f16 with 2-cell spatial halo (zero) and channel pad (zero).
__global__ void pack_in_kernel(const float* __restrict__ x, _Float16* __restrict__ out,
                               int B, int C, int Cp, int H, int W)
{
  const int Hp = H + 4, Wp = W + 4;
  int i = blockIdx.x * blockDim.x + threadIdx.x;
  int tot = B * Hp * Wp * Cp;
  if (i >= tot) return;
  int c = i % Cp; int t = i / Cp;
  int w = t % Wp; t /= Wp;
  int h = t % Hp; int b = t / Hp;
  int hh = h - 2, ww = w - 2;
  float v = 0.0f;
  if (c < C && (unsigned)hh < (unsigned)H && (unsigned)ww < (unsigned)W)
    v = x[((size_t)(b * C + c) * H + hh) * W + ww];
  out[i] = (_Float16)v;
}

// Zero the 2-cell halo of a padded NHWC buffer at its current logical shape.
__global__ void halo_zero_kernel(_Float16* __restrict__ buf, int B, int H, int W, int Cp)
{
  const int Hp = H + 4, Wp = W + 4;
  int i = blockIdx.x * blockDim.x + threadIdx.x;
  int tot = B * Hp * Wp * Cp;
  if (i >= tot) return;
  int t = i / Cp;
  int w = t % Wp; t /= Wp;
  int h = t % Hp;
  if (h >= 2 && h < H + 2 && w >= 2 && w < W + 2) return;   // interior untouched
  buf[i] = (_Float16)0.0f;
}

// ---------------------------------------------------------------------------
// Implicit-GEMM conv on v_wmma_f32_16x16x32_f16, NHWC f16 (+halo) activations.
// Per wave: 32x32 output tile (2 oc-tiles x 2 sp-tiles) -> 4 WMMAs per K-step
// fed by 8 b128 loads (2 loads/WMMA). Nested kh/kw tap loops with incremental
// pointer stepping (no per-tap divisions); software-pipelined K loop with
// peeled final iteration; all loads unconditional (halo + clamped tails).
// ---------------------------------------------------------------------------
template<int KH, int KW>
__global__ __launch_bounds__(128)
void conv_nhwc_wmma(const _Float16* __restrict__ act,  // NHWC Cinp (+halo if haloIn)
                    const _Float16* __restrict__ wp,   // packed weights
                    const float* __restrict__ bias,
                    const _Float16* __restrict__ res,  // NHWC resCp with halo, or null
                    const float* __restrict__ badd,    // (B,Cout) f32, or null
                    _Float16* __restrict__ outh,       // NHWC outCp (+halo if haloOut)
                    float* __restrict__ outf,          // NCHW f32 (mode f32)
                    int NB, int Cinp, int H, int Wd, int Cout, int pad,
                    int outCp, int cOff, int resCp, int relu, int haloIn, int haloOut)
{
  const int S = H * Wd;
  const int Ntot = NB * S;
  const int KT = Cinp >> 5;
  const int T = KH * KW;
  const int Mtiles = (Cout + 15) >> 4;   // always even in this network
  const int inHp = H + 4 * haloIn, inWp = Wd + 4 * haloIn, hoI = 2 * haloIn;
  const int lane = threadIdx.x & 31;
  const int wid  = threadIdx.x >> 5;
  const int half = lane >> 4;
  const int l16  = lane & 15;
  const int mtb  = blockIdx.x * 2;
  const int spb  = (blockIdx.y * 4 + wid) * 32;
  if (spb >= Ntot || mtb >= Mtiles) return;   // wave-uniform

  // two 16-column sets per wave
  int  bbv[2], posv[2], ohv[2], owv[2];
  bool nval[2];
#pragma unroll
  for (int ni = 0; ni < 2; ++ni) {
    int n = spb + ni * 16 + l16;
    nval[ni] = (n < Ntot);
    int nc = nval[ni] ? n : (Ntot - 1);
    bbv[ni] = nc / S;
    posv[ni] = nc - bbv[ni] * S;
    ohv[ni] = posv[ni] / Wd;
    owv[ni] = posv[ni] - ohv[ni] * Wd;
  }

  v8f acc00 = {}, acc01 = {}, acc10 = {}, acc11 = {};
  const size_t aStride = (size_t)T * KT * 512;   // halfs between oc-tiles
  const size_t aTap    = (size_t)KT * 512;       // halfs per tap
  const size_t rowStep = (size_t)inWp * Cinp;    // halfs per input row

  // base pointers at tap (kh=0, kw=0); step incrementally over taps
  const _Float16* bbase0 = act +
      ((size_t)(bbv[0] * inHp + (ohv[0] - pad + hoI)) * inWp + (owv[0] - pad + hoI)) * Cinp
      + half * 16;
  const _Float16* bbase1 = act +
      ((size_t)(bbv[1] * inHp + (ohv[1] - pad + hoI)) * inWp + (owv[1] - pad + hoI)) * Cinp
      + half * 16;
  const _Float16* arow = wp + ((size_t)(mtb * T * KT) * 32 + lane) * 16;

#pragma unroll 1
  for (int kh = 0; kh < KH; ++kh) {
    const _Float16* brow0 = bbase0;
    const _Float16* brow1 = bbase1;
#pragma unroll 1
    for (int kw = 0; kw < KW; ++kw) {
      // software pipeline with peeled final iteration
      v16h b0c = ld16(brow0);
      v16h b1c = ld16(brow1);
      v16h a0c = ld16(arow);
      v16h a1c = ld16(arow + aStride);
      for (int kt = 0; kt < KT - 1; ++kt) {
        v16h b0n = ld16(brow0 + (size_t)(kt + 1) * 32);
        v16h b1n = ld16(brow1 + (size_t)(kt + 1) * 32);
        v16h a0n = ld16(arow + (size_t)(kt + 1) * 512);
        v16h a1n = ld16(arow + (size_t)(kt + 1) * 512 + aStride);
        acc00 = __builtin_amdgcn_wmma_f32_16x16x32_f16(false, a0c, false, b0c, (short)0, acc00, false, false);
        acc10 = __builtin_amdgcn_wmma_f32_16x16x32_f16(false, a1c, false, b0c, (short)0, acc10, false, false);
        acc01 = __builtin_amdgcn_wmma_f32_16x16x32_f16(false, a0c, false, b1c, (short)0, acc01, false, false);
        acc11 = __builtin_amdgcn_wmma_f32_16x16x32_f16(false, a1c, false, b1c, (short)0, acc11, false, false);
        b0c = b0n; b1c = b1n; a0c = a0n; a1c = a1n;
      }
      acc00 = __builtin_amdgcn_wmma_f32_16x16x32_f16(false, a0c, false, b0c, (short)0, acc00, false, false);
      acc10 = __builtin_amdgcn_wmma_f32_16x16x32_f16(false, a1c, false, b0c, (short)0, acc10, false, false);
      acc01 = __builtin_amdgcn_wmma_f32_16x16x32_f16(false, a0c, false, b1c, (short)0, acc01, false, false);
      acc11 = __builtin_amdgcn_wmma_f32_16x16x32_f16(false, a1c, false, b1c, (short)0, acc11, false, false);
      brow0 += Cinp; brow1 += Cinp; arow += aTap;   // next tap (kw+1)
    }
    bbase0 += rowStep; bbase1 += rowStep;           // next tap row (kh+1)
  }

#pragma unroll
  for (int ni = 0; ni < 2; ++ni) {
    if (!nval[ni]) continue;   // per-lane; fine after all WMMAs
    const int bb = bbv[ni], pos = posv[ni], oh = ohv[ni], ow = owv[ni];
#pragma unroll
    for (int mi = 0; mi < 2; ++mi) {
      const v8f* accp = mi ? (ni ? &acc11 : &acc10) : (ni ? &acc01 : &acc00);
      const int c0 = (mtb + mi) * 16 + 8 * half;   // 8 consecutive channels
      if (outf) {
#pragma unroll
        for (int r = 0; r < 8; ++r) {
          int oc2 = c0 + r;
          if (oc2 < Cout) {
            float v = (*accp)[r] + bias[oc2];
            if (relu) v = (v > 0.0f) ? v : 0.0f;
            if (badd) v += badd[bb * Cout + oc2];
            outf[((size_t)bb * Cout + oc2) * S + pos] = v;
          }
        }
      } else {
        const size_t p3 = (size_t)(bb * (H + 4 * haloOut) + oh + 2 * haloOut)
                          * (Wd + 4 * haloOut) + ow + 2 * haloOut;
        const size_t pr = (size_t)(bb * (H + 4) + oh + 2) * (Wd + 4) + ow + 2;
        v8h ov;
#pragma unroll
        for (int r = 0; r < 8; ++r) {
          int oc2 = c0 + r;
          float v = 0.0f;
          if (oc2 < Cout) {
            v = (*accp)[r] + bias[oc2];
            if (relu) v = (v > 0.0f) ? v : 0.0f;
            if (res)  v += (float)res[pr * resCp + oc2];
            if (badd) v += badd[bb * Cout + oc2];
          }
          ov[r] = (_Float16)v;
        }
        *(v8h*)(outh + p3 * outCp + cOff + c0) = ov;
      }
    }
  }
}

// --------------------- elementwise helpers (NHWC f16 + halo) ---------------

__global__ void maxpool_kernel(const _Float16* __restrict__ in, _Float16* __restrict__ out,
                               int B, int C, int H, int W, int OH, int OW)
{
  const int Hp = H + 4, Wp = W + 4, OHp = OH + 4, OWp = OW + 4;
  int i = blockIdx.x * blockDim.x + threadIdx.x;
  int tot = B * OH * OW * C;
  if (i >= tot) return;
  int c = i % C; int t = i / C;
  int ow = t % OW; t /= OW;
  int oh = t % OH; int b = t / OH;
  int hs = oh * 2, ws = ow * 2;
  float m = -__builtin_inff();
  for (int kh = 0; kh < 3; ++kh) {
    int ih = hs + kh; if (ih >= H) break;
    for (int kw = 0; kw < 3; ++kw) {
      int iw = ws + kw; if (iw >= W) break;
      float v = (float)in[((size_t)(b * Hp + ih + 2) * Wp + iw + 2) * C + c];
      m = (v > m) ? v : m;
    }
  }
  out[((size_t)(b * OHp + oh + 2) * OWp + ow + 2) * C + c] = (_Float16)m;
}

__global__ void upsample_kernel(const _Float16* __restrict__ in, _Float16* __restrict__ out,
                                int B, int C, int H, int W, int OH, int OW)
{
  const int Hp = H + 4, Wp = W + 4, OHp = OH + 4, OWp = OW + 4;
  int i = blockIdx.x * blockDim.x + threadIdx.x;
  int tot = B * OH * OW * C;
  if (i >= tot) return;
  int c = i % C; int t = i / C;
  int ow = t % OW; t /= OW;
  int oh = t % OH; int b = t / OH;
  int ih = (oh * H) / OH;
  int iw = (ow * W) / OW;
  out[((size_t)(b * OHp + oh + 2) * OWp + ow + 2) * C + c] =
      in[((size_t)(b * Hp + ih + 2) * Wp + iw + 2) * C + c];
}

// Fill channels [192,224) of a 224-channel halo'd NHWC buffer (interior only).
__global__ void writey_kernel(const float* __restrict__ y, _Float16* __restrict__ out,
                              int B, int H, int W)
{
  const int Hp = H + 4, Wp = W + 4;
  int i = blockIdx.x * blockDim.x + threadIdx.x;
  int tot = B * H * W * 32;
  if (i >= tot) return;
  int c = i % 32; int t = i / 32;
  int w = t % W; t /= W;
  int h = t % H; int b = t / H;
  out[((size_t)(b * Hp + h + 2) * Wp + w + 2) * 224 + 192 + c] =
      (c < 18) ? (_Float16)y[b * 18 + c] : (_Float16)0.0f;
}

__global__ void copy_kernel(const float* __restrict__ src, float* __restrict__ dst, int n)
{
  int i = blockIdx.x * blockDim.x + threadIdx.x;
  if (i < n) dst[i] = src[i];
}

__global__ void zero_kernel(float* __restrict__ p, int n)
{
  int i = blockIdx.x * blockDim.x + threadIdx.x;
  if (i < n) p[i] = 0.0f;
}

// Superpixel mean, labels/channels 0..2 of f32 NCHW x9 (in d_out).
__global__ void sp_accum_kernel(const int* __restrict__ sp, const float* __restrict__ x9,
                                float* __restrict__ acc, int B, int S)
{
  int i = blockIdx.x * blockDim.x + threadIdx.x;
  if (i >= B * S) return;
  int s = i % S; int b = i / S;
  int t = sp[i];
  if ((unsigned)t < 3u) {
    atomicAdd(&acc[b * 3 + t], 1.0f);
    for (int c = 0; c < 3; ++c)
      atomicAdd(&acc[24 + (b * 3 + c) * 3 + t], x9[((size_t)b * 18 + c) * S + s]);
  }
}

__global__ void sp_apply_kernel(const int* __restrict__ sp, float* __restrict__ x9,
                                const float* __restrict__ acc, int B, int S)
{
  int i = blockIdx.x * blockDim.x + threadIdx.x;
  int tot = B * 3 * S;
  if (i >= tot) return;
  int s = i % S; int t2 = i / S;
  int c = t2 % 3; int b = t2 / 3;
  int t = sp[b * S + s];
  if ((unsigned)t < 3u)
    x9[((size_t)b * 18 + c) * S + s] = acc[24 + (b * 3 + c) * 3 + t] / acc[b * 3 + t];
}

// ------------------------------- host side ---------------------------------

static inline void launch_conv(hipStream_t st, const _Float16* act, const _Float16* wp,
                               const float* bias, const _Float16* res, const float* badd,
                               _Float16* outh, float* outf,
                               int NB, int Cinp, int H, int W, int Cout, int Ksz, int pad,
                               int outCp, int cOff, int resCp, int relu, int hin, int hout)
{
  dim3 grid(CDIV(Cout, 16) / 2, CDIV(NB * H * W, 128));
  dim3 block(128);
  if (Ksz == 5)
    conv_nhwc_wmma<5, 5><<<grid, block, 0, st>>>(act, wp, bias, res, badd, outh, outf,
                                                 NB, Cinp, H, W, Cout, pad, outCp, cOff, resCp, relu, hin, hout);
  else if (Ksz == 3)
    conv_nhwc_wmma<3, 3><<<grid, block, 0, st>>>(act, wp, bias, res, badd, outh, outf,
                                                 NB, Cinp, H, W, Cout, pad, outCp, cOff, resCp, relu, hin, hout);
  else
    conv_nhwc_wmma<1, 1><<<grid, block, 0, st>>>(act, wp, bias, res, badd, outh, outf,
                                                 NB, Cinp, H, W, Cout, pad, outCp, cOff, resCp, relu, hin, hout);
}

extern "C" void kernel_launch(void* const* d_in, const int* in_sizes, int n_in,
                              void* d_out, int out_size, void* d_ws, size_t ws_size,
                              hipStream_t stream)
{
  (void)in_sizes; (void)n_in; (void)out_size; (void)ws_size;

  const float* x  = (const float*)d_in[0];   // (8,3,150,100)
  const int*   sp = (const int*)d_in[1];     // (8,150,100)
  const float* W[20], *Bi[20];
  for (int k = 0; k < 9; ++k) { W[k] = (const float*)d_in[2 + 2 * k]; Bi[k] = (const float*)d_in[3 + 2 * k]; }
  const float* fc1w = (const float*)d_in[20]; const float* fc1b = (const float*)d_in[21];
  const float* fc2w = (const float*)d_in[22]; const float* fc2b = (const float*)d_in[23];
  for (int k = 9; k < 19; ++k) { W[k] = (const float*)d_in[24 + 2 * (k - 9)]; Bi[k] = (const float*)d_in[25 + 2 * (k - 9)]; }

  const int B = 8;
  const int S1 = 150 * 100;
  const size_t S1p = 154 * 104, S2p = 79 * 54, S3p = 41 * 29, S4p = 22 * 16;

  // -------- arena --------
  char* ws = (char*)d_ws;
  size_t off = 0;
  auto takeB = [&](size_t bytes) { size_t r = off; off = (off + bytes + 255) & ~(size_t)255; return r; };
  auto takeH = [&](size_t halfs) { return (_Float16*)(ws + takeB(halfs * 2)); };
  auto takeF = [&](size_t fl)    { return (float*)(ws + takeB(fl * 4)); };

  _Float16* Xh   = takeH((size_t)B * S1p * 32);
  _Float16* A_x1 = takeH((size_t)B * S1p * 192);
  _Float16* A_x2 = takeH((size_t)B * S2p * 192);
  _Float16* A_x3 = takeH((size_t)B * S3p * 192);
  _Float16* A_x4 = takeH((size_t)B * S4p * 192);
  _Float16* A_yc = takeH((size_t)B * 216 * 96);     // no halo (FC flatten)
  _Float16* A_y1 = takeH((size_t)B * 1024);
  float*    A_y  = takeF(B * 18);
  float*    A_ac = takeF(128);
  _Float16* Sa   = takeH((size_t)B * S1p * 192);
  _Float16* Sb   = takeH((size_t)B * S1p * 256);
  _Float16* Sc   = takeH((size_t)B * S1p * 192);

  auto pack = [&](const float* wt, int Cout, int Cin, int T, int permSx, int permCx) {
    int Cinp = (Cin + 31) & ~31;
    size_t n = (size_t)CDIV(Cout, 16) * T * (Cinp / 32) * 512;
    _Float16* out = takeH(n);
    pack_w_kernel<<<CDIV((int)n, 256), 256, 0, stream>>>(wt, out, Cout, Cin, T, Cinp,
                                                         permSx, permCx, (int)n);
    return out;
  };
  _Float16* P1  = pack(W[0],  128, 3,   25, 0, 0);
  _Float16* P2  = pack(W[1],  192, 128, 25, 0, 0);
  _Float16* P3  = pack(W[2],  192, 192, 25, 0, 0);
  _Float16* P4  = pack(W[3],  192, 192, 25, 0, 0);
  _Float16* P5  = pack(W[4],  192, 192, 25, 0, 0);
  _Float16* P6  = pack(W[5],  192, 192, 25, 0, 0);
  _Float16* P7  = pack(W[6],  192, 192, 25, 0, 0);
  _Float16* P8  = pack(W[7],  192, 192, 25, 0, 0);
  _Float16* P9  = pack(W[8],  96,  192, 1,  0, 0);
  _Float16* Pf1 = pack(fc1w, 1024, 216 * 96, 1, 216, 96);
  _Float16* Pf2 = pack(fc2w,   18, 1024, 1, 0, 0);
  _Float16* P10 = pack(W[9],  192, 192, 25, 0, 0);
  _Float16* P11 = pack(W[10], 192, 210, 25, 0, 0);
  _Float16* P12 = pack(W[11], 192, 192, 9,  0, 0);
  _Float16* P13 = pack(W[12], 192, 210, 25, 0, 0);
  _Float16* P14 = pack(W[13], 192, 192, 25, 0, 0);
  _Float16* P15 = pack(W[14], 192, 210, 25, 0, 0);
  _Float16* P16 = pack(W[15], 192, 3,   9,  0, 0);
  _Float16* P17 = pack(W[16], 256, 192, 9,  0, 0);
  _Float16* P18 = pack(W[17], 18,  256, 1,  0, 0);
  _Float16* P19 = pack(W[18], 18,  18,  1,  0, 0);

  float* x9 = (float*)d_out;
  float* y_out = (float*)d_out + (size_t)B * 18 * S1;

  pack_in_kernel<<<CDIV((int)(B * S1p * 32), 256), 256, 0, stream>>>(x, Xh, B, 3, 32, 150, 100);

  auto hz = [&](_Float16* buf, int H, int Wd, int Cp) {
    int tot = B * (H + 4) * (Wd + 4) * Cp;
    halo_zero_kernel<<<CDIV(tot, 256), 256, 0, stream>>>(buf, B, H, Wd, Cp);
  };
  auto ew = [&](void (*k)(const _Float16*, _Float16*, int, int, int, int, int, int),
                const _Float16* in, _Float16* out, int C, int H, int Wd, int OH, int OW) {
    int tot = B * OH * OW * C;
    k<<<CDIV(tot, 256), 256, 0, stream>>>(in, out, B, C, H, Wd, OH, OW);
  };

  // -------- encoder --------
  launch_conv(stream, Xh, P1, Bi[0], nullptr, nullptr, Sa, nullptr, B, 32, 150, 100, 128, 5, 2, 128, 0, 0, 1, 1, 1);
  hz(Sa, 150, 100, 128);
  launch_conv(stream, Sa, P2, Bi[1], nullptr, nullptr, A_x1, nullptr, B, 128, 150, 100, 192, 5, 2, 192, 0, 0, 1, 1, 1);
  ew(maxpool_kernel, A_x1, Sa, 192, 150, 100, 75, 50);  hz(Sa, 75, 50, 192);
  launch_conv(stream, Sa, P3, Bi[2], nullptr, nullptr, Sb, nullptr, B, 192, 75, 50, 192, 5, 2, 192, 0, 0, 1, 1, 1);
  hz(Sb, 75, 50, 192);
  launch_conv(stream, Sb, P4, Bi[3], nullptr, nullptr, A_x2, nullptr, B, 192, 75, 50, 192, 5, 2, 192, 0, 0, 1, 1, 1);
  ew(maxpool_kernel, A_x2, Sa, 192, 75, 50, 37, 25);    hz(Sa, 37, 25, 192);
  launch_conv(stream, Sa, P5, Bi[4], nullptr, nullptr, Sb, nullptr, B, 192, 37, 25, 192, 5, 2, 192, 0, 0, 1, 1, 1);
  hz(Sb, 37, 25, 192);
  launch_conv(stream, Sb, P6, Bi[5], nullptr, nullptr, A_x3, nullptr, B, 192, 37, 25, 192, 5, 2, 192, 0, 0, 1, 1, 1);
  ew(maxpool_kernel, A_x3, Sa, 192, 37, 25, 18, 12);    hz(Sa, 18, 12, 192);
  launch_conv(stream, Sa, P7, Bi[6], nullptr, nullptr, Sb, nullptr, B, 192, 18, 12, 192, 5, 2, 192, 0, 0, 1, 1, 1);
  hz(Sb, 18, 12, 192);
  launch_conv(stream, Sb, P8, Bi[7], nullptr, nullptr, A_x4, nullptr, B, 192, 18, 12, 192, 5, 2, 192, 0, 0, 1, 1, 1);

  // -------- head (yc has no halo; FC flatten is contiguous per batch) ------
  launch_conv(stream, A_x4, P9, Bi[8], nullptr, nullptr, A_yc, nullptr, B, 192, 18, 12, 96, 1, 0, 96, 0, 0, 1, 1, 0);
  launch_conv(stream, A_yc, Pf1, fc1b, nullptr, nullptr, A_y1, nullptr, B, 216 * 96, 1, 1, 1024, 1, 0, 1024, 0, 0, 1, 0, 0);
  launch_conv(stream, A_y1, Pf2, fc2b, nullptr, nullptr, nullptr, A_y, B, 1024, 1, 1, 18, 1, 0, 0, 0, 0, 1, 0, 0);

  // -------- decoder (concat buffers: NHWC Cp=224, y in channels 192..209) --
  ew(upsample_kernel, A_x4, Sa, 192, 18, 12, 37, 25);   hz(Sa, 37, 25, 192);
  launch_conv(stream, Sa, P10, Bi[9], A_x3, nullptr, Sb, nullptr, B, 192, 37, 25, 192, 5, 2, 224, 0, 192, 1, 1, 1);
  writey_kernel<<<CDIV(B * 37 * 25 * 32, 256), 256, 0, stream>>>(A_y, Sb, B, 37, 25);
  hz(Sb, 37, 25, 224);
  launch_conv(stream, Sb, P11, Bi[10], nullptr, nullptr, Sc, nullptr, B, 224, 37, 25, 192, 5, 2, 192, 0, 0, 1, 1, 1);
  ew(upsample_kernel, Sc, Sa, 192, 37, 25, 75, 50);     hz(Sa, 75, 50, 192);
  launch_conv(stream, Sa, P12, Bi[11], A_x2, nullptr, Sb, nullptr, B, 192, 75, 50, 192, 3, 1, 224, 0, 192, 1, 1, 1);
  writey_kernel<<<CDIV(B * 75 * 50 * 32, 256), 256, 0, stream>>>(A_y, Sb, B, 75, 50);
  hz(Sb, 75, 50, 224);
  launch_conv(stream, Sb, P13, Bi[12], nullptr, nullptr, Sc, nullptr, B, 224, 75, 50, 192, 5, 2, 192, 0, 0, 1, 1, 1);
  ew(upsample_kernel, Sc, Sa, 192, 75, 50, 150, 100);   hz(Sa, 150, 100, 192);
  launch_conv(stream, Sa, P14, Bi[13], A_x1, nullptr, Sb, nullptr, B, 192, 150, 100, 192, 5, 2, 224, 0, 192, 1, 1, 1);
  writey_kernel<<<CDIV(B * S1 * 32, 256), 256, 0, stream>>>(A_y, Sb, B, 150, 100);
  hz(Sb, 150, 100, 224);

  // x8 = r(conv(x7c,w15)) + r(conv(x,w16)); then w17
  launch_conv(stream, Xh, P16, Bi[15], nullptr, nullptr, Sc, nullptr, B, 32, 150, 100, 192, 3, 1, 192, 0, 0, 1, 1, 1);
  launch_conv(stream, Sb, P15, Bi[14], Sc, nullptr, Sa, nullptr, B, 224, 150, 100, 192, 5, 2, 192, 0, 192, 1, 1, 1);
  hz(Sa, 150, 100, 192);
  launch_conv(stream, Sa, P17, Bi[16], nullptr, nullptr, Sb, nullptr, B, 192, 150, 100, 256, 3, 1, 256, 0, 0, 1, 1, 1);

  // x9 = conv(conv(x8b,w18) + y_bcast, w19)   (final layer f32 NCHW)
  launch_conv(stream, Sb, P18, Bi[17], nullptr, A_y, Sc, nullptr, B, 256, 150, 100, 18, 1, 0, 32, 0, 0, 0, 1, 1);
  launch_conv(stream, Sc, P19, Bi[18], nullptr, nullptr, nullptr, x9, B, 32, 150, 100, 18, 1, 0, 0, 0, 0, 0, 1, 0);

  // -------- superpixel means + y output --------
  zero_kernel<<<1, 128, 0, stream>>>(A_ac, 128);
  sp_accum_kernel<<<CDIV(B * S1, 256), 256, 0, stream>>>(sp, x9, A_ac, B, S1);
  sp_apply_kernel<<<CDIV(B * 3 * S1, 256), 256, 0, stream>>>(sp, x9, A_ac, B, S1);
  copy_kernel<<<1, 256, 0, stream>>>(A_y, y_out, B * 18);
}
